// SynthesisBlock_63995012711096
// MI455X (gfx1250) — compile-verified, matrix operand order checked by reading, not run on hardware
//
#include <hip/hip_runtime.h>

typedef __attribute__((ext_vector_type(16))) _Float16 v16h;
typedef __attribute__((ext_vector_type(8)))  _Float16 v8h;
typedef __attribute__((ext_vector_type(8)))  float    v8f;

#define B_     4
#define CIN    64
#define COUT   64
#define R_     32
#define R3     (R_*R_*R_)          // 32768
#define WDIM   512
#define TAPS   27

// ---------------------------------------------------------------------------
// Kernel 0: styles[b][ci] = w_latent[b] . (affine_w[ci] / sqrt(512)) + affine_b[ci]
// ---------------------------------------------------------------------------
__global__ void styles_k(const float* __restrict__ wl,
                         const float* __restrict__ aw,
                         const float* __restrict__ ab,
                         float* __restrict__ styles) {
    const int t  = threadIdx.x;          // 256 threads: (b, ci)
    const int b  = t >> 6;
    const int ci = t & 63;
    const float g = 0.04419417382415922f;  // 1/sqrt(512)
    float s = ab[ci];
    for (int k = 0; k < WDIM; ++k)
        s += wl[b * WDIM + k] * aw[ci * WDIM + k] * g;
    styles[b * CIN + ci] = s;
}

// ---------------------------------------------------------------------------
// Kernel 1: modulate + demodulate weights, pack as f16 in WMMA A-fragment order.
// A-fragment (16x32 f16, ISA 7.12.2): lane = hi*16 + m (m = co&15),
//   half index h in 0..15 maps to K:  h<8 -> K = hi*8 + h ; h>=8 -> K = 16 + hi*8 + (h-8)
// Flat layout: apack[((((b*4+co_tile)*27+tap)*2+cchunk)*32+lane)*16 + h]
// ---------------------------------------------------------------------------
__global__ void modpack(const float* __restrict__ weight,   // [COUT][CIN][27]
                        const float* __restrict__ styles,   // [B][CIN]
                        _Float16*    __restrict__ apack) {
    const int b  = blockIdx.x >> 6;
    const int co = blockIdx.x & 63;
    __shared__ float red[128];
    __shared__ float dsh;

    float ss = 0.f;
    for (int e = threadIdx.x; e < CIN * TAPS; e += 128) {
        const int ci = e / TAPS, tap = e % TAPS;
        const float wv = weight[(co * CIN + ci) * TAPS + tap] * styles[b * CIN + ci];
        ss += wv * wv;
    }
    red[threadIdx.x] = ss;
    __syncthreads();
    for (int s = 64; s > 0; s >>= 1) {
        if (threadIdx.x < s) red[threadIdx.x] += red[threadIdx.x + s];
        __syncthreads();
    }
    if (threadIdx.x == 0) dsh = rsqrtf(red[0] + 1e-8f);
    __syncthreads();
    const float d = dsh;

    const int co_tile = co >> 4;
    const int m       = co & 15;
    for (int e = threadIdx.x; e < CIN * TAPS; e += 128) {
        const int ci = e / TAPS, tap = e % TAPS;
        const float wv = weight[(co * CIN + ci) * TAPS + tap] * styles[b * CIN + ci] * d;
        const int cchunk = ci >> 5;
        const int cl     = ci & 31;
        const int hi     = (cl >> 3) & 1;
        const int h      = (cl & 7) + ((cl >> 4) << 3);
        const int lane   = hi * 16 + m;
        apack[((((b * 4 + co_tile) * TAPS + tap) * 2 + cchunk) * 32 + lane) * 16 + h] =
            (_Float16)wv;
    }
}

// ---------------------------------------------------------------------------
// Kernel 2: implicit-GEMM conv via v_wmma_f32_16x16x32_f16.
// Workgroup = 128 threads (4 waves); owns (b, 16 consecutive w positions).
// x halo tile staged in LDS as f16, ci-fastest -> B fragments are 2x ds_load_b128.
// ---------------------------------------------------------------------------
__global__ void __launch_bounds__(128)
conv_wmma(const float*    __restrict__ x,       // [B][CIN][R][R][R]
          const _Float16* __restrict__ apack,
          const float*    __restrict__ noise,   // [B][1][R][R][R]
          const float*    __restrict__ bias,    // [COUT]
          float*          __restrict__ out) {   // [B][COUT][R][R][R]
    const int b    = blockIdx.x >> 11;          // 2048 tiles per batch
    const int tile = blockIdx.x & 2047;
    const int p0   = tile << 4;                 // 16 positions along w
    const int d0   = p0 >> 10;
    const int h0   = (p0 >> 5) & 31;
    const int w0   = p0 & 31;                   // 0 or 16

    // LDS x tile: [rel_d(3)][rel_h(3)][rel_w(18)][ci(64)] f16  = 10368 halves
    __shared__ alignas(16) _Float16 xlds[3 * 3 * 18 * CIN];

    const int tid = threadIdx.x;
    // 10368 = 81 * 128 : exact, no tail divergence
    for (int it = 0; it < 81; ++it) {
        const int e  = tid + it * 128;
        const int ci = e / 162;                 // 162 = 3*3*18
        const int r  = e % 162;
        const int rd = r / 54;
        const int r2 = r % 54;
        const int rh = r2 / 18;
        const int rw = r2 % 18;
        const int gd = d0 + rd - 1;
        const int gh = h0 + rh - 1;
        const int gw = w0 + rw - 1;
        float v = 0.f;
        if ((unsigned)gd < 32u && (unsigned)gh < 32u && (unsigned)gw < 32u)
            v = x[(((b * CIN + ci) * R_ + gd) * R_ + gh) * R_ + gw];
        xlds[r * CIN + ci] = (_Float16)v;
    }
    __syncthreads();

    const int wave = tid >> 5;                  // COUT tile 0..3
    const int lane = tid & 31;
    const int n    = lane & 15;                 // position within tile
    const int hi   = lane >> 4;

    v8f acc = {};
    const _Float16* abase = apack + ((size_t)(b * 4 + wave)) * TAPS * 2 * 32 * 16;

    for (int kd = 0; kd < 3; ++kd) {
        for (int kh = 0; kh < 3; ++kh) {
            const _Float16* xrow = &xlds[(kd * 3 + kh) * 18 * CIN];
#pragma unroll
            for (int kw = 0; kw < 3; ++kw) {
                const int tap = (kd * 3 + kh) * 3 + kw;
#pragma unroll
                for (int cc = 0; cc < 2; ++cc) {
                    // A fragment: one contiguous 32B load per lane
                    const v16h a =
                        *(const v16h*)(abase + (((tap * 2 + cc) * 32 + lane) << 4));
                    // B fragment: two 16B LDS loads (the two 8-ci runs this lane needs)
                    const _Float16* xs = xrow + (n + kw) * CIN + cc * 32 + hi * 8;
                    const v8h blo = *(const v8h*)(xs);
                    const v8h bhi = *(const v8h*)(xs + 16);
                    const v16h bf = __builtin_shufflevector(
                        blo, bhi, 0, 1, 2, 3, 4, 5, 6, 7, 8, 9, 10, 11, 12, 13, 14, 15);
                    acc = __builtin_amdgcn_wmma_f32_16x16x32_f16(
                        false, a, false, bf, (short)0, acc, false, false);
                }
            }
        }
    }

    // Epilogue: C/D layout -> element (M = hi*8 + j, N = n)
    const float nz = noise[b * R3 + p0 + n];
#pragma unroll
    for (int j = 0; j < 8; ++j) {
        const int co = wave * 16 + hi * 8 + j;
        float y = acc[j] + nz + bias[co];
        y = (y < 0.f ? 0.2f * y : y) * 1.41421356237f;   // lrelu * sqrt(2)
        out[((size_t)(b * COUT + co)) * R3 + p0 + n] = y;
    }
}

// ---------------------------------------------------------------------------
extern "C" void kernel_launch(void* const* d_in, const int* in_sizes, int n_in,
                              void* d_out, int out_size, void* d_ws, size_t ws_size,
                              hipStream_t stream) {
    const float* x      = (const float*)d_in[0];
    const float* wl     = (const float*)d_in[1];
    const float* weight = (const float*)d_in[2];
    const float* aw     = (const float*)d_in[3];
    const float* ab     = (const float*)d_in[4];
    const float* noise  = (const float*)d_in[5];
    const float* bias   = (const float*)d_in[6];
    float* out = (float*)d_out;

    float*    styles = (float*)d_ws;                       // 256 f32 = 1KB
    _Float16* apack  = (_Float16*)((char*)d_ws + 1024);    // 4*4*27*2*32*16 f16 = 864KB

    styles_k<<<1, 256, 0, stream>>>(wl, aw, ab, styles);
    modpack<<<B_ * COUT, 128, 0, stream>>>(weight, styles, apack);
    conv_wmma<<<B_ * (R3 / 16), 128, 0, stream>>>(x, apack, noise, bias, out);
}
